// QuantumNet_36756330119802
// MI455X (gfx1250) — matrix operand order repaired and checked by reading, not compiled, not run on hardware
//
#include <hip/hip_runtime.h>

// MI455X / gfx1250, wave32. Fused: GEMM(65536x512 @ 512x4, WMMA f16->f32)
// -> tanh -> 4-qubit statevector circuit (registers) -> 4x2 projection.
// Memory-bound: one 128MB streaming pass of input_features.

typedef __attribute__((ext_vector_type(16))) _Float16 v16h;
typedef __attribute__((ext_vector_type(8)))  _Float16 v8h;
typedef __attribute__((ext_vector_type(8)))  float    v8f;

#define IN_DIM   512
#define N_QUBITS 4
#define Q_DEPTH  6
#define LDS_STRIDE 520   // halves; 1040B row stride, 16B aligned, bank-spread

// RY(theta): new0 = c*a - s*b ; new1 = s*a + c*b   (pair split by MASK bit)
template<int MASK>
__device__ __forceinline__ void ry_gate(float st[16], float ch, float sh) {
#pragma unroll
  for (int i = 0; i < 16; ++i) {
    if ((i & MASK) == 0) {
      float a = st[i], b = st[i | MASK];
      st[i]        = ch * a - sh * b;
      st[i | MASK] = sh * a + ch * b;
    }
  }
}

// CNOT: where control bit set, swap target-bit pair
template<int CM, int TM>
__device__ __forceinline__ void cnot_gate(float st[16]) {
#pragma unroll
  for (int i = 0; i < 16; ++i) {
    if ((i & CM) != 0 && (i & TM) == 0) {
      float t = st[i]; st[i] = st[i | TM]; st[i | TM] = t;
    }
  }
}

__global__ __launch_bounds__(256)
void qnet_fused_kernel(const float* __restrict__ x,
                       const float* __restrict__ pre_w,
                       const float* __restrict__ pre_b,
                       const float* __restrict__ qp,
                       const float* __restrict__ post_w,
                       const float* __restrict__ post_b,
                       float* __restrict__ out) {
  __shared__ __align__(16) _Float16 wlds[16][LDS_STRIDE];

  const int tid = threadIdx.x;

  // ---- Stage pre_w as f16 into LDS, rows 4..15 zero (so WMMA A rows pad to 0)
  {
    unsigned int* p = (unsigned int*)&wlds[0][0];
#pragma unroll 1
    for (int i = tid; i < 16 * LDS_STRIDE / 2; i += 256) p[i] = 0u;
  }
  __syncthreads();
#pragma unroll 1
  for (int i = tid; i < N_QUBITS * IN_DIM; i += 256)
    wlds[i >> 9][i & 511] = (_Float16)pre_w[i];
  __syncthreads();

  const int lane    = tid & 31;
  const int wave    = tid >> 5;
  const int rowBase = (blockIdx.x * 8 + wave) * 16;   // 16 samples per wave
  const int m       = lane & 15;                      // sample / weight row
  const int hi      = lane >> 4;                      // lane-half select

  const float* rowp = x + (size_t)(rowBase + m) * IN_DIM;

  v8f acc = {0.f, 0.f, 0.f, 0.f, 0.f, 0.f, 0.f, 0.f};

  // ---- K loop: 16 steps of V_WMMA_F32_16X16X32_F16 covering K=512
#pragma unroll 4
  for (int step = 0; step < 16; ++step) {
    const int k0 = step * 32;

    // A (pre_w, f16) from LDS in contiguous k order (same as B):
    //   lane<16: K=k0+0..15 ; lane>=16: K=k0+16..31
    union { v16h v; v8h h[2]; } A;
    A.h[0] = *(const v8h*)&wlds[m][k0 + hi * 16];
    A.h[1] = *(const v8h*)&wlds[m][k0 + hi * 16 + 8];

    // B (input^T) in contiguous k order: lane<16 holds K=k0+0..15 of its row,
    // lane>=16 holds K=k0+16..31 -> 64B f32 contiguous per lane, convert to f16.
    const float* bp = rowp + k0 + hi * 16;
    float4 f0 = *(const float4*)(bp + 0);
    float4 f1 = *(const float4*)(bp + 4);
    float4 f2 = *(const float4*)(bp + 8);
    float4 f3 = *(const float4*)(bp + 12);
    if (step + 4 < 16) __builtin_prefetch(bp + 128, 0, 1);

    v16h B;
    B[0]  = (_Float16)f0.x; B[1]  = (_Float16)f0.y;
    B[2]  = (_Float16)f0.z; B[3]  = (_Float16)f0.w;
    B[4]  = (_Float16)f1.x; B[5]  = (_Float16)f1.y;
    B[6]  = (_Float16)f1.z; B[7]  = (_Float16)f1.w;
    B[8]  = (_Float16)f2.x; B[9]  = (_Float16)f2.y;
    B[10] = (_Float16)f2.z; B[11] = (_Float16)f2.w;
    B[12] = (_Float16)f3.x; B[13] = (_Float16)f3.y;
    B[14] = (_Float16)f3.z; B[15] = (_Float16)f3.w;

    // D = A*B + C : C[r][m] = sum_k pre_w[r,k] * x[rowBase+m, k]
    acc = __builtin_amdgcn_wmma_f32_16x16x32_f16(
        /*neg_a=*/false, A.v, /*neg_b=*/false, B,
        /*c_mod=*/(short)0, acc, /*reuse_a=*/false, /*reuse_b=*/false);
  }

  // ---- Per-sample quantum circuit: lane m (0..15) owns sample rowBase+m.
  // C layout: VGPR r, lanes 0-15 => M=r, N=lane -> acc[r] = channel r.
  if (lane < 16) {
    const int sample = rowBase + lane;

    float st[16];
#pragma unroll
    for (int i = 0; i < 16; ++i) st[i] = 0.25f;   // H^4 |0000>, real

    // q_in = tanh(pre + b) * pi/2 ; RY uses theta/2 -> fold: * pi/4
#pragma unroll
    for (int r = 0; r < 4; ++r) {
      float h = tanhf(acc[r] + pre_b[r]) * 0.78539816339744831f;
      float sh = __sinf(h), ch = __cosf(h);
      if (r == 0) ry_gate<8>(st, ch, sh);
      if (r == 1) ry_gate<4>(st, ch, sh);
      if (r == 2) ry_gate<2>(st, ch, sh);
      if (r == 3) ry_gate<1>(st, ch, sh);
    }

#pragma unroll
    for (int k = 0; k < Q_DEPTH; ++k) {
      cnot_gate<8, 4>(st);   // CNOT(0,1)
      cnot_gate<2, 1>(st);   // CNOT(2,3)
      cnot_gate<4, 2>(st);   // CNOT(1,2)
      float h0 = qp[k * 4 + 0] * 0.5f; ry_gate<8>(st, __cosf(h0), __sinf(h0));
      float h1 = qp[k * 4 + 1] * 0.5f; ry_gate<4>(st, __cosf(h1), __sinf(h1));
      float h2 = qp[k * 4 + 2] * 0.5f; ry_gate<2>(st, __cosf(h2), __sinf(h2));
      float h3 = qp[k * 4 + 3] * 0.5f; ry_gate<1>(st, __cosf(h3), __sinf(h3));
    }

    // <Z_q> = sum probs(bit q = 0) - sum probs(bit q = 1)
    float z0 = 0.f, z1 = 0.f, z2 = 0.f, z3 = 0.f;
#pragma unroll
    for (int i = 0; i < 16; ++i) {
      float p = st[i] * st[i];
      z0 += (i & 8) ? -p : p;
      z1 += (i & 4) ? -p : p;
      z2 += (i & 2) ? -p : p;
      z3 += (i & 1) ? -p : p;
    }

    float o0 = post_b[0] + z0 * post_w[0] + z1 * post_w[1]
                         + z2 * post_w[2] + z3 * post_w[3];
    float o1 = post_b[1] + z0 * post_w[4] + z1 * post_w[5]
                         + z2 * post_w[6] + z3 * post_w[7];
    *(float2*)(out + (size_t)sample * 2) = make_float2(o0, o1);
  }
}

extern "C" void kernel_launch(void* const* d_in, const int* in_sizes, int n_in,
                              void* d_out, int out_size, void* d_ws, size_t ws_size,
                              hipStream_t stream) {
  (void)in_sizes; (void)n_in; (void)d_ws; (void)ws_size; (void)out_size;
  const float* x      = (const float*)d_in[0];  // [65536,512]
  const float* pre_w  = (const float*)d_in[1];  // [4,512]
  const float* pre_b  = (const float*)d_in[2];  // [4]
  const float* qp     = (const float*)d_in[3];  // [24]
  const float* post_w = (const float*)d_in[4];  // [2,4]
  const float* post_b = (const float*)d_in[5];  // [2]
  float* out = (float*)d_out;                   // [65536,2]

  // 256 threads = 8 wave32, 16 samples/wave -> 128 samples/block -> 512 blocks
  hipLaunchKernelGGL(qnet_fused_kernel, dim3(512), dim3(256), 0, stream,
                     x, pre_w, pre_b, qp, post_w, post_b, out);
}
